// SlaterJastrowComplex_26912265077269
// MI455X (gfx1250) — matrix operand order repaired and checked by reading, not compiled, not run on hardware
//
#include <hip/hip_runtime.h>
#include <stdint.h>

// ---------------------------------------------------------------------------
// Types
// ---------------------------------------------------------------------------
typedef _Float16 v16h __attribute__((ext_vector_type(16)));
typedef float    v8f  __attribute__((ext_vector_type(8)));
typedef unsigned int u32x4 __attribute__((ext_vector_type(4)));
typedef int          i32x4 __attribute__((ext_vector_type(4)));
typedef int          i32x8 __attribute__((ext_vector_type(8)));

__device__ __forceinline__ unsigned ballot32(bool p) {
#if __has_builtin(__builtin_amdgcn_ballot_w32)
    return __builtin_amdgcn_ballot_w32(p);
#else
    return (unsigned)__ballot(p);
#endif
}

// ---------------------------------------------------------------------------
// Kernel B: Jastrow via WMMA.  J = n @ [k_re | k_im | 0...0]  (16x16 C tile,
// 16 samples per wave, K=128 in 4 chunks of 32).  n is exactly 0/1 in f16.
// Writes out[s] = {tanh(|J_s|), 0}.
// ---------------------------------------------------------------------------
__global__ __launch_bounds__(256) void jastrow_wmma_kernel(
    const float* __restrict__ n, const float* __restrict__ kc,
    float2* __restrict__ out)
{
    const int lane = threadIdx.x & 31;
    const int wv   = threadIdx.x >> 5;
    const int tile = blockIdx.x * 8 + wv;
    const int s0   = tile * 16;
    const int m    = lane & 15;
    const bool hi  = lane >= 16;

    const float* nrow = n + (size_t)(s0 + m) * 128;

    // Lane-column selectors as register masks (avoid if-conversion to
    // exec-masked loads: loads below are unconditional float4 vectors).
    const float selR = (m == 0) ? 1.0f : 0.0f;
    const float selI = (m == 1) ? 1.0f : 0.0f;

    v8f c = {};
#pragma unroll
    for (int q = 0; q < 4; ++q) {
        const int kb = q * 32;
        // A fragment (contiguous K order, same as B): lanes<16 carry K = kb+[0..15];
        // lanes>=16 carry K = kb+[16..31].
        const int base1 = kb + (hi ? 16 : 0);
        const int base2 = base1 + 8;
        float4 r0 = *(const float4*)(nrow + base1);
        float4 r1 = *(const float4*)(nrow + base1 + 4);
        float4 r2 = *(const float4*)(nrow + base2);
        float4 r3 = *(const float4*)(nrow + base2 + 4);
        v16h a;
        a[0]=(_Float16)r0.x; a[1]=(_Float16)r0.y; a[2]=(_Float16)r0.z; a[3]=(_Float16)r0.w;
        a[4]=(_Float16)r1.x; a[5]=(_Float16)r1.y; a[6]=(_Float16)r1.z; a[7]=(_Float16)r1.w;
        a[8]=(_Float16)r2.x; a[9]=(_Float16)r2.y; a[10]=(_Float16)r2.z; a[11]=(_Float16)r2.w;
        a[12]=(_Float16)r3.x; a[13]=(_Float16)r3.y; a[14]=(_Float16)r3.z; a[15]=(_Float16)r3.w;

        // B fragment (32x16): N = lane&15; only columns 0 (k_re) and 1 (k_im)
        // are non-zero.  lanes<16 carry K = kb+[0..15]; lanes>=16 kb+[16..31].
        // kc is interleaved (re,im): one float4 covers two K values.
        const int kO = kb + (hi ? 16 : 0);
        v16h b;
#pragma unroll
        for (int t = 0; t < 8; ++t) {
            float4 kv = *(const float4*)(kc + 2 * kO + 4 * t);  // K = kO+2t, kO+2t+1
            float v0 = fmaf(selR, kv.x, selI * kv.y);
            float v1 = fmaf(selR, kv.z, selI * kv.w);
            b[2 * t]     = (_Float16)v0;
            b[2 * t + 1] = (_Float16)v1;
        }
        c = __builtin_amdgcn_wmma_f32_16x16x32_f16(false, a, false, b,
                                                   (short)0, c, false, false);
    }

    // C layout: Jr (col 0) lives in lanes 0/16, Ji (col 1) in lanes 1/17.
    float t[8];
#pragma unroll
    for (int v = 0; v < 8; ++v) {
        float ji = __shfl(c[v], (lane & 16) | 1);
        float jr = c[v];
        t[v] = tanhf(sqrtf(fmaf(jr, jr, ji * ji)));
    }
    if (m == 0) {
        const int mb = hi ? 8 : 0;
#pragma unroll
        for (int v = 0; v < 8; ++v)
            out[s0 + mb + v] = make_float2(t[v], 0.0f);
    }
}

// ---------------------------------------------------------------------------
// Kernel A: per-sample complex log-dets.  One wave per sample; lanes 0-15 =
// spin-0 rows, lanes 16-31 = spin-1 rows.  Virtual partial pivoting (no row
// swaps; parity from ballot inversion count).  Accumulates into d_out.
// ---------------------------------------------------------------------------
__global__ __launch_bounds__(256) void slater_logdet_kernel(
    const float* __restrict__ n, const float* __restrict__ M,
    float2* __restrict__ out)
{
    __shared__ float ldsM[2048];   // 64 rows x 16 cols x {re,im}  = 8 KB

#if __has_builtin(__builtin_amdgcn_tensor_load_to_lds)
    if (threadIdx.x < 32) {
        const unsigned lds_base = (unsigned)(uintptr_t)&ldsM[0];
        const unsigned long long ga = (unsigned long long)(uintptr_t)M;
        // D# group0: count=1 | lds_addr | global_addr | type=2
        u32x4 g0 = { 1u, lds_base, (unsigned)ga,
                     (unsigned)((ga >> 32) & 0x1FFFFFFu) | 0x80000000u };
        // D# group1: data_size=4B; tensor_dim0=2048, tensor_dim1=1,
        // tile_dim0=2048, tensor_dim0_stride=2048 (1-D copy of 8 KB).
        i32x8 g1 = { 0x00020000, 0x08000000, 0x00010000, 0x08000000,
                     0, 2048, 0, 0 };
        i32x4 z4 = { 0, 0, 0, 0 };
#if defined(__clang_major__) && (__clang_major__ >= 23)
        i32x8 z8 = { 0, 0, 0, 0, 0, 0, 0, 0 };
        __builtin_amdgcn_tensor_load_to_lds(g0, g1, z4, z4, z8, 0);
#else
        __builtin_amdgcn_tensor_load_to_lds(g0, g1, z4, z4, 0);
#endif
#if __has_builtin(__builtin_amdgcn_s_wait_tensorcnt)
        __builtin_amdgcn_s_wait_tensorcnt(0);
#else
        asm volatile("s_wait_tensorcnt 0x0" ::: "memory");
#endif
    }
    __syncthreads();
#else
    for (int i = threadIdx.x; i < 2048; i += 256) ldsM[i] = M[i];
    __syncthreads();
#endif

    const int lane   = threadIdx.x & 31;
    const int wv     = threadIdx.x >> 5;
    const int sample = blockIdx.x * 8 + wv;
    const float* nrow = n + (size_t)sample * 128;

    // Occupation masks via ballots over the 0/1 floats.
    unsigned b0 = ballot32(nrow[lane]      > 0.5f);
    unsigned b1 = ballot32(nrow[32 + lane] > 0.5f);
    unsigned b2 = ballot32(nrow[64 + lane] > 0.5f);
    unsigned b3 = ballot32(nrow[96 + lane] > 0.5f);
    unsigned long long m0 = ((unsigned long long)b1 << 32) | b0;
    unsigned long long m1 = ((unsigned long long)b3 << 32) | b2;

    const int r    = lane & 15;
    const int spin = lane >> 4;
    unsigned long long mm = spin ? m1 : m0;
    for (int t = 0; t < r; ++t) mm &= (mm - 1ull);   // drop r lowest set bits
    const int idx = __ffsll(mm) - 1;                 // r-th occupied orbital

    // Gather my row of M (16 complex = 128 B) from LDS as 8 x b128.
    float ar[16], ai[16];
    const float4* row4 = (const float4*)&ldsM[idx * 32];
#pragma unroll
    for (int t = 0; t < 8; ++t) {
        float4 v = row4[t];
        ar[2 * t]     = v.x;  ai[2 * t]     = v.y;
        ar[2 * t + 1] = v.z;  ai[2 * t + 1] = v.w;
    }

    float logabs = 0.0f, phr = 1.0f, phi = 0.0f;
    int   invs = 0;
    bool  done = false;

#pragma unroll
    for (int k = 0; k < 16; ++k) {
        // ---- pivot search: arg-max of |a[k]|^2 over not-yet-pivoted lanes
        float cand = done ? -1.0f : fmaf(ar[k], ar[k], ai[k] * ai[k]);
        int   ci   = lane;
#pragma unroll
        for (int off = 8; off; off >>= 1) {
            float ov = __shfl_xor(cand, off);
            int   oi = __shfl_xor(ci,   off);
            bool take = (ov > cand) || (ov == cand && oi < ci);
            cand = take ? ov : cand;
            ci   = take ? oi : ci;
        }
        const int p = ci;  // absolute pivot lane within this half

        // ---- permutation parity: previously-done lanes above p
        unsigned dm  = ballot32(done);
        unsigned hdm = spin ? (dm >> 16) : (dm & 0xFFFFu);
        invs += __popc((hdm & 0xFFFFu) >> ((p & 15) + 1));

        // ---- pivot value broadcast, accumulate log|u| and unit phase
        float urk = __shfl(ar[k], p);
        float uik = __shfl(ai[k], p);
        float d2  = fmaf(urk, urk, uik * uik);
        logabs += 0.5f * __logf(d2);
        float invm = rsqrtf(d2);
        float npr = (phr * urk - phi * uik) * invm;
        float npi = (phr * uik + phi * urk) * invm;
        phr = npr; phi = npi;

        // ---- rank-1 elimination (predicated, shuffles stay uniform)
        const bool doneNow = done || (lane == p);
        float invd2 = 1.0f / d2;
        float fr = (ar[k] * urk + ai[k] * uik) * invd2;
        float fi = (ai[k] * urk - ar[k] * uik) * invd2;
#pragma unroll
        for (int j = k + 1; j < 16; ++j) {
            float urj = __shfl(ar[j], p);
            float uij = __shfl(ai[j], p);
            float nar = ar[j] - (fr * urj - fi * uij);
            float nai = ai[j] - (fr * uij + fi * urj);
            ar[j] = doneNow ? ar[j] : nar;
            ai[j] = doneNow ? ai[j] : nai;
        }
        done = doneNow;
    }

    if (invs & 1) { phr = -phr; phi = -phi; }
    float im = atan2f(phi, phr);     // principal arg per spin (matches slogdet)

    // Combine the two spin halves.
    float o_re = __shfl(logabs, lane ^ 16);
    float o_im = __shfl(im,     lane ^ 16);
    float tre = logabs + o_re;
    float tim = im + o_im;

    if (lane == 0) {
        float2 cur = out[sample];                 // jastrow written by kernel B
        out[sample] = make_float2(cur.x + tre, cur.y + tim);
    }
}

// ---------------------------------------------------------------------------
// Launch
// ---------------------------------------------------------------------------
extern "C" void kernel_launch(void* const* d_in, const int* in_sizes, int n_in,
                              void* d_out, int out_size, void* d_ws, size_t ws_size,
                              hipStream_t stream) {
    const float* n  = (const float*)d_in[0];   // (B,128) float32 0/1
    const float* M  = (const float*)d_in[1];   // (64,16) complex64 -> float pairs
    const float* kc = (const float*)d_in[2];   // (128,)  complex64 -> float pairs
    float2* out = (float2*)d_out;              // (B,) complex64 -> float2

    const int batch = in_sizes[0] / 128;       // 65536

    // Jastrow first (pure write of d_out), then log-dets accumulate into it.
    jastrow_wmma_kernel<<<batch / 16 / 8, 256, 0, stream>>>(n, kc, out);
    slater_logdet_kernel<<<batch / 8, 256, 0, stream>>>(n, M, out);
}
